// Net_81020263072365
// MI455X (gfx1250) — compile-verified, matrix operand order checked by reading, not run on hardware
//
#include <hip/hip_runtime.h>
#include <hip/hip_bf16.h>

typedef __attribute__((ext_vector_type(16))) _Float16 v16h;
typedef __attribute__((ext_vector_type(4)))  _Float16 v4h;
typedef __attribute__((ext_vector_type(8)))  float    v8f;

#define NNODES 98304
#define NPM    24
#define NMONO  4096
#define NGRAPH 2048
#define NEDGE  393216
#define FIN    64
#define HDIM   128

// ---------------------------------------------------------------- utilities

__global__ void zero_f32_kernel(float* __restrict__ p, long n4) {
    long t = (long)blockIdx.x * blockDim.x + threadIdx.x;
    if (t < n4) ((float4*)p)[t] = make_float4(0.f, 0.f, 0.f, 0.f);
}

// Wt[o*K + k] = (f16) W[k*HOUT + o]   (convert + transpose)
__global__ void wt_convert_kernel(const float* __restrict__ W, _Float16* __restrict__ Wt,
                                  int K, int HOUT) {
    int t = blockIdx.x * blockDim.x + threadIdx.x;
    if (t >= K * HOUT) return;
    int k = t / HOUT, o = t % HOUT;
    Wt[(size_t)o * K + k] = (_Float16)W[(size_t)k * HOUT + o];
}

// agg[dst[e]][f..f+3] += h[src[e]][f..f+3]   (agg is L2-resident: 50MB << 192MB L2)
__global__ void scatter_add_kernel(const float* __restrict__ h, const int* __restrict__ src,
                                   const int* __restrict__ dst, float* __restrict__ agg,
                                   int C) {
    long t = (long)blockIdx.x * blockDim.x + threadIdx.x;
    int grp = C >> 2;
    long total = (long)NEDGE * grp;
    if (t >= total) return;
    int e = (int)(t / grp);
    int f = (int)(t % grp) << 2;
    int s = src[e], d = dst[e];
    float4 v = *(const float4*)(h + (size_t)s * C + f);
    float* o = agg + (size_t)d * C + f;
    atomicAdd(o + 0, v.x);
    atomicAdd(o + 1, v.y);
    atomicAdd(o + 2, v.z);
    atomicAdd(o + 3, v.w);
}

// a16 = (f16)(h + agg)   -> WMMA A operand
__global__ void add_convert_kernel(const float* __restrict__ h, const float* __restrict__ agg,
                                   _Float16* __restrict__ out, long n4) {
    long t = (long)blockIdx.x * blockDim.x + threadIdx.x;
    if (t >= n4) return;
    float4 a = ((const float4*)h)[t];
    float4 b = ((const float4*)agg)[t];
    v4h r;
    r.x = (_Float16)(a.x + b.x);
    r.y = (_Float16)(a.y + b.y);
    r.z = (_Float16)(a.z + b.z);
    r.w = (_Float16)(a.w + b.w);
    ((v4h*)out)[t] = r;
}

// ------------------------------------------------------- WMMA GEMM (the core)
// C[row, 0..127] = relu(A[row, 0..K-1] @ Wt^T + bias)
//   A  : [Nrows, K]  f16 row-major
//   Wt : [128, K]    f16 (transposed weights -> contiguous B fragments)
// One block = 8 waves = 64 rows x 128 cols.
//   wave w owns columns [16w, 16w+16); it loads its K/32 B fragments ONCE into
//   registers, then iterates 4 row tiles of 16, issuing K/32 v_wmma per tile.
// A fragment: lane l -> row (l&15), K-halves (l>>4)*16 .. +15  (contiguous v16h)
// B fragment: lane l -> col (l&15), K-halves (l>>4)*16 .. +15  (contiguous v16h of Wt)
// D layout:   lane l -> col (l&15), acc[v] = row v + 8*(l>>4)
template <int K, bool WRITE_F32, bool WRITE_F16>
__global__ __launch_bounds__(256) void gemm_wmma_kernel(
    const _Float16* __restrict__ A, const _Float16* __restrict__ Wt,
    const float* __restrict__ bias, float* __restrict__ Cf,
    _Float16* __restrict__ Ch) {
    constexpr int KB = K / 32;
    const int lane = threadIdx.x & 31;
    const int wave = threadIdx.x >> 5;       // 0..7 -> column tile
    const int l15  = lane & 15;
    const int half = lane >> 4;              // 0 | 1
    const int row0 = blockIdx.x << 6;        // 64 rows per block
    const int col0 = wave << 4;
    const int col  = col0 + l15;

    // B fragments: resident for the whole block's 64 rows
    v16h bfrag[KB];
    const _Float16* bP = Wt + (size_t)(col0 + l15) * K + half * 16;
#pragma unroll
    for (int i = 0; i < KB; ++i) bfrag[i] = *(const v16h*)(bP + i * 32);

    const float bb = bias[col];

#pragma unroll
    for (int r = 0; r < 4; ++r) {
        const int rbase = row0 + (r << 4);
        const _Float16* aP = A + (size_t)(rbase + l15) * K + half * 16;
        v8f acc = {};
#pragma unroll
        for (int i = 0; i < KB; ++i) {
            v16h a = *(const v16h*)(aP + i * 32);
            acc = __builtin_amdgcn_wmma_f32_16x16x32_f16(
                /*neg_a=*/false, a, /*neg_b=*/false, bfrag[i],
                /*c_mod=*/(short)0, acc, /*reuse_a=*/false, /*reuse_b=*/false);
        }
#pragma unroll
        for (int v = 0; v < 8; ++v) {
            int row = rbase + v + (half << 3);
            float val = fmaxf(acc[v] + bb, 0.f);
            size_t idx = (size_t)row * HDIM + col;
            if (WRITE_F32) Cf[idx] = val;
            if (WRITE_F16) Ch[idx] = (_Float16)val;
        }
    }
}

// ------------------------------------------------------- pooling / attention

// block = monomer (128 threads = feature); emb = mean of 24 node rows, frac = min ratio
__global__ __launch_bounds__(128) void pool_kernel(const float* __restrict__ h,
                                                   const float* __restrict__ ratio,
                                                   float* __restrict__ emb,
                                                   float* __restrict__ frac) {
    int m = blockIdx.x, f = threadIdx.x;
    int base = m * NPM;
    float s = 0.f;
#pragma unroll 4
    for (int i = 0; i < NPM; ++i) s += h[(size_t)(base + i) * HDIM + f];
    emb[(size_t)m * HDIM + f] = s * (1.0f / NPM);
    if (f == 0) {
        float mn = ratio[base];
        for (int i = 1; i < NPM; ++i) mn = fminf(mn, ratio[base + i]);
        frac[m] = mn;
    }
}

// block = monomer; q=(emb@wq+bq)*frac, k=(emb@wk+bk)*frac, v=emb@wv+bv
__global__ __launch_bounds__(128) void qkv_kernel(
    const float* __restrict__ emb, const float* __restrict__ frac,
    const float* __restrict__ wq, const float* __restrict__ bq,
    const float* __restrict__ wk, const float* __restrict__ bk,
    const float* __restrict__ wv, const float* __restrict__ bv,
    float* __restrict__ q, float* __restrict__ k, float* __restrict__ v) {
    __shared__ float e[HDIM];
    int m = blockIdx.x, f = threadIdx.x;
    e[f] = emb[(size_t)m * HDIM + f];
    __syncthreads();
    float aq = bq[f], ak = bk[f], av = bv[f];
#pragma unroll 4
    for (int i = 0; i < HDIM; ++i) {
        float x = e[i];
        aq = fmaf(x, wq[i * HDIM + f], aq);
        ak = fmaf(x, wk[i * HDIM + f], ak);
        av = fmaf(x, wv[i * HDIM + f], av);
    }
    float fr = frac[m];
    size_t idx = (size_t)m * HDIM + f;
    q[idx] = aq * fr;
    k[idx] = ak * fr;
    v[idx] = av;
}

// block = graph (2 monomers): per-channel 2-way softmax + weighted sum of v
__global__ __launch_bounds__(128) void attn_kernel(const float* __restrict__ q,
                                                   const float* __restrict__ k,
                                                   const float* __restrict__ v,
                                                   float* __restrict__ polymer) {
    int g = blockIdx.x, f = threadIdx.x;
    size_t i0 = (size_t)(2 * g) * HDIM + f;
    size_t i1 = (size_t)(2 * g + 1) * HDIM + f;
    const float scale = 0.08838834764831845f;   // 1/sqrt(128)
    float ks = k[i0] + k[i1];
    float e0 = q[i0] * ks * scale;
    float e1 = q[i1] * ks * scale;
    float mx = fmaxf(e0, e1);
    float x0 = __expf(e0 - mx), x1 = __expf(e1 - mx);
    polymer[(size_t)g * HDIM + f] = (v[i0] * x0 + v[i1] * x1) / (x0 + x1);
}

// block = graph: out[g] = relu(polymer@wo1+bo1) @ wo2 + bo2
__global__ __launch_bounds__(128) void head_kernel(const float* __restrict__ polymer,
                                                   const float* __restrict__ wo1,
                                                   const float* __restrict__ bo1,
                                                   const float* __restrict__ wo2,
                                                   const float* __restrict__ bo2,
                                                   float* __restrict__ out) {
    __shared__ float p[HDIM];
    __shared__ float red[HDIM];
    int g = blockIdx.x, f = threadIdx.x;
    p[f] = polymer[(size_t)g * HDIM + f];
    __syncthreads();
    float a = bo1[f];
#pragma unroll 4
    for (int i = 0; i < HDIM; ++i) a = fmaf(p[i], wo1[i * HDIM + f], a);
    a = fmaxf(a, 0.f);
    red[f] = a * wo2[f];
    __syncthreads();
    for (int s = 64; s > 0; s >>= 1) {
        if (f < s) red[f] += red[f + s];
        __syncthreads();
    }
    if (f == 0) out[g] = red[0] + bo2[0];
}

// ---------------------------------------------------------------- host side

static inline char* wsalloc(char*& p, size_t bytes) {
    char* r = p;
    p += (bytes + 255) & ~(size_t)255;
    return r;
}

extern "C" void kernel_launch(void* const* d_in, const int* in_sizes, int n_in,
                              void* d_out, int out_size, void* d_ws, size_t ws_size,
                              hipStream_t stream) {
    const float* x       = (const float*)d_in[0];
    const float* ratio   = (const float*)d_in[1];
    const int*   src     = (const int*)d_in[4];           // edge_index[0]
    const int*   dst     = ((const int*)d_in[4]) + NEDGE; // edge_index[1]
    const float* w1[3] = {(const float*)d_in[6],  (const float*)d_in[10], (const float*)d_in[14]};
    const float* b1[3] = {(const float*)d_in[7],  (const float*)d_in[11], (const float*)d_in[15]};
    const float* w2[3] = {(const float*)d_in[8],  (const float*)d_in[12], (const float*)d_in[16]};
    const float* b2[3] = {(const float*)d_in[9],  (const float*)d_in[13], (const float*)d_in[17]};
    const float* wq = (const float*)d_in[18]; const float* bq = (const float*)d_in[19];
    const float* wk = (const float*)d_in[20]; const float* bk = (const float*)d_in[21];
    const float* wv = (const float*)d_in[22]; const float* bv = (const float*)d_in[23];
    const float* wo1 = (const float*)d_in[24]; const float* bo1 = (const float*)d_in[25];
    const float* wo2 = (const float*)d_in[26]; const float* bo2 = (const float*)d_in[27];
    float* out = (float*)d_out;

    // ---- workspace partition
    char* wp = (char*)d_ws;
    float*    agg  = (float*)   wsalloc(wp, (size_t)NNODES * HDIM * 4);
    _Float16* a16  = (_Float16*)wsalloc(wp, (size_t)NNODES * HDIM * 2);
    _Float16* t16  = (_Float16*)wsalloc(wp, (size_t)NNODES * HDIM * 2);
    float*    hA   = (float*)   wsalloc(wp, (size_t)NNODES * HDIM * 4);
    float*    hB   = (float*)   wsalloc(wp, (size_t)NNODES * HDIM * 4);
    _Float16* wt1[3]; _Float16* wt2[3];
    wt1[0] = (_Float16*)wsalloc(wp, (size_t)HDIM * FIN * 2);
    wt2[0] = (_Float16*)wsalloc(wp, (size_t)HDIM * HDIM * 2);
    wt1[1] = (_Float16*)wsalloc(wp, (size_t)HDIM * HDIM * 2);
    wt2[1] = (_Float16*)wsalloc(wp, (size_t)HDIM * HDIM * 2);
    wt1[2] = (_Float16*)wsalloc(wp, (size_t)HDIM * HDIM * 2);
    wt2[2] = (_Float16*)wsalloc(wp, (size_t)HDIM * HDIM * 2);
    float* emb     = (float*)wsalloc(wp, (size_t)NMONO * HDIM * 4);
    float* frac    = (float*)wsalloc(wp, (size_t)NMONO * 4);
    float* qb      = (float*)wsalloc(wp, (size_t)NMONO * HDIM * 4);
    float* kb_     = (float*)wsalloc(wp, (size_t)NMONO * HDIM * 4);
    float* vb      = (float*)wsalloc(wp, (size_t)NMONO * HDIM * 4);
    float* polymer = (float*)wsalloc(wp, (size_t)NGRAPH * HDIM * 4);

    const int T = 256;

    // ---- weight conversion (f16 + transpose)
    wt_convert_kernel<<<(FIN * HDIM + T - 1) / T, T, 0, stream>>>(w1[0], wt1[0], FIN, HDIM);
    wt_convert_kernel<<<(HDIM * HDIM + T - 1) / T, T, 0, stream>>>(w2[0], wt2[0], HDIM, HDIM);
    for (int l = 1; l < 3; ++l) {
        wt_convert_kernel<<<(HDIM * HDIM + T - 1) / T, T, 0, stream>>>(w1[l], wt1[l], HDIM, HDIM);
        wt_convert_kernel<<<(HDIM * HDIM + T - 1) / T, T, 0, stream>>>(w2[l], wt2[l], HDIM, HDIM);
    }

    const int gemmGrid = NNODES / 64;   // 1536 blocks, 8 waves, 64x128 out each

    // ---- GIN layer 0 (input features FIN=64)
    {
        long n = (long)NNODES * FIN;
        zero_f32_kernel<<<(unsigned)((n / 4 + T - 1) / T), T, 0, stream>>>(agg, n / 4);
        long st = (long)NEDGE * (FIN / 4);
        scatter_add_kernel<<<(unsigned)((st + T - 1) / T), T, 0, stream>>>(x, src, dst, agg, FIN);
        add_convert_kernel<<<(unsigned)((n / 4 + T - 1) / T), T, 0, stream>>>(x, agg, a16, n / 4);
        gemm_wmma_kernel<FIN, false, true><<<gemmGrid, 256, 0, stream>>>(a16, wt1[0], b1[0], nullptr, t16);
        gemm_wmma_kernel<HDIM, true, false><<<gemmGrid, 256, 0, stream>>>(t16, wt2[0], b2[0], hA, nullptr);
    }

    // ---- GIN layers 1,2 (H=128), ping-pong hA <-> hB
    const float* hin = hA;
    float* hout = hB;
    for (int l = 1; l < 3; ++l) {
        long n = (long)NNODES * HDIM;
        zero_f32_kernel<<<(unsigned)((n / 4 + T - 1) / T), T, 0, stream>>>(agg, n / 4);
        long st = (long)NEDGE * (HDIM / 4);
        scatter_add_kernel<<<(unsigned)((st + T - 1) / T), T, 0, stream>>>(hin, src, dst, agg, HDIM);
        add_convert_kernel<<<(unsigned)((n / 4 + T - 1) / T), T, 0, stream>>>(hin, agg, a16, n / 4);
        gemm_wmma_kernel<HDIM, false, true><<<gemmGrid, 256, 0, stream>>>(a16, wt1[l], b1[l], nullptr, t16);
        gemm_wmma_kernel<HDIM, true, false><<<gemmGrid, 256, 0, stream>>>(t16, wt2[l], b2[l], hout, nullptr);
        const float* tmp = hin; hin = hout; hout = (float*)tmp;
    }
    const float* hfin = hin;   // output of layer 2

    // ---- pooling + attention + head
    pool_kernel<<<NMONO, 128, 0, stream>>>(hfin, ratio, emb, frac);
    qkv_kernel<<<NMONO, 128, 0, stream>>>(emb, frac, wq, bq, wk, bk, wv, bv, qb, kb_, vb);
    attn_kernel<<<NGRAPH, 128, 0, stream>>>(qb, kb_, vb, polymer);
    head_kernel<<<NGRAPH, 128, 0, stream>>>(polymer, wo1, bo1, wo2, bo2, out);
}
